// GeometricBEVLifter_79611513798994
// MI455X (gfx1250) — matrix operand order repaired and checked by reading, not compiled, not run on hardware
//
#include <hip/hip_runtime.h>
#include <hip/hip_bf16.h>
#include <math.h>

typedef __attribute__((ext_vector_type(16))) _Float16 v16h;
typedef __attribute__((ext_vector_type(8)))  _Float16 v8h;
typedef __attribute__((ext_vector_type(8)))  float    v8f;

#define BB   2
#define VV   6
#define NBV  12          // B*V
#define FCH  64
#define HFd  28
#define WFd  50
#define PP   (HFd*WFd)   // 1400
#define PT16 ((PP + 15) / 16)  // 88 pixel tiles
#define BCH  128
#define BH   128
#define BW   128
#define ND   32
#define D_MIN_ 1.0f
#define D_MAX_ 50.0f
#define EXT_   20.0f
#define EPS_   1e-5f

__device__ __forceinline__ float gelu_f(float x) {
  return 0.5f * x * (1.0f + erff(x * 0.70710678118654752f));
}
__device__ __forceinline__ float bn_f(float x, float g, float be, float m, float v) {
  return (x - m) * (g * rsqrtf(v + EPS_)) + be;
}

// ---------------- feat_maps f32 NCHW -> f16 [bv][p][ci]  (B-matrix feed, 128B pixel stride)
__global__ void k_fm_nhwc(const float* __restrict__ src, _Float16* __restrict__ dst) {
  int tid = blockIdx.x * blockDim.x + threadIdx.x;
  if (tid >= NBV * PP * FCH) return;
  int ci = tid % FCH;
  int p  = (tid / FCH) % PP;
  int bv = tid / (FCH * PP);
  dst[tid] = (_Float16)src[((size_t)bv * FCH + ci) * PP + p];
}

// ---------------- weight repack: f32 [co][ci][3][3] -> f16 [tap][co][ci] (generic channels)
__global__ void k_wcvt(const float* __restrict__ w, _Float16* __restrict__ wa,
                       int nco, int nci) {
  int tid = blockIdx.x * blockDim.x + threadIdx.x;
  if (tid >= 9 * nco * nci) return;
  int ci  = tid % nci;
  int co  = (tid / nci) % nco;
  int tap = tid / (nci * nco);
  int ky = tap / 3, kx = tap % 3;
  wa[tid] = (_Float16)w[(((size_t)co * nci + ci) * 3 + ky) * 3 + kx];
}

// ---------------- depth head conv1 (3x3, 64->64) + BN + GELU via WMMA
// One wave: 16 out-ch x 16 pixels; K = 9 taps x 64 ci (2 chunks of 32). 18 WMMAs/tile.
__global__ void k_depth_conv1_wmma(const _Float16* __restrict__ fmh, const _Float16* __restrict__ wd,
                                   const float* __restrict__ b,  const float* __restrict__ g,
                                   const float* __restrict__ be, const float* __restrict__ m,
                                   const float* __restrict__ v,  float* __restrict__ h) {
  int wave = (blockIdx.x * blockDim.x + threadIdx.x) >> 5;   // uniform per wave
  int lane = threadIdx.x & 31;
  if (wave >= NBV * 4 * PT16) return;
  int pt   = wave % PT16;
  int co_t = (wave / PT16) & 3;
  int bv   = wave / (PT16 * 4);
  int hl = lane >> 4, n = lane & 15;
  int p = pt * 16 + n;
  bool pvalid = p < PP;
  int x = p % WFd, y = p / WFd;            // only consumed when pvalid

  v8f acc = {};
  const _Float16* inb = fmh + (size_t)bv * PP * FCH;

  #pragma unroll
  for (int ky = 0; ky < 3; ++ky) {
    int iy = y + ky - 1;
    #pragma unroll
    for (int kx = 0; kx < 3; ++kx) {
      int ix = x + kx - 1;
      bool bvalid = pvalid && (iy >= 0) && (iy < HFd) && (ix >= 0) && (ix < WFd);
      const _Float16* bp = inb + ((size_t)iy * WFd + ix) * FCH;
      int tap = ky * 3 + kx;
      const _Float16* ap = wd + ((size_t)tap * FCH + (co_t * 16 + n)) * FCH;
      #pragma unroll
      for (int ch = 0; ch < 2; ++ch) {
        v16h bmat = {};
        if (bvalid) bmat = *(const v16h*)(bp + ch * 32 + 16 * hl);
        union { v16h vv; v8h h2[2]; } am;
        am.h2[0] = *(const v8h*)(ap + ch * 32 + 16 * hl);
        am.h2[1] = *(const v8h*)(ap + ch * 32 + 16 * hl + 8);
        acc = __builtin_amdgcn_wmma_f32_16x16x32_f16(
            false, am.vv, false, bmat, (short)0, acc, false, false);
      }
    }
  }
  if (pvalid) {
    #pragma unroll
    for (int j = 0; j < 8; ++j) {
      int co = co_t * 16 + hl * 8 + j;
      float val = acc[j] + b[co];
      val = gelu_f(bn_f(val, g[co], be[co], m[co], v[co]));
      h[((size_t)bv * FCH + co) * PP + p] = val;   // f32 [bv][co][p] for softmax stage
    }
  }
}

// ---------------- depth head conv2 (1x1, 64->32) + softmax over depth
__global__ void k_depth_softmax(const float* __restrict__ h, const float* __restrict__ w2,
                                const float* __restrict__ b2, float* __restrict__ dp) {
  int tid = blockIdx.x * blockDim.x + threadIdx.x;
  if (tid >= NBV * PP) return;
  int p = tid % PP, bv = tid / PP;
  float lg[ND];
  #pragma unroll
  for (int d = 0; d < ND; ++d) lg[d] = b2[d];
  const float* hb = h + (size_t)bv * FCH * PP + p;
  for (int ci = 0; ci < FCH; ++ci) {
    float hv = hb[(size_t)ci * PP];
    #pragma unroll
    for (int d = 0; d < ND; ++d) lg[d] = fmaf(w2[d * FCH + ci], hv, lg[d]);
  }
  float mx = lg[0];
  #pragma unroll
  for (int d = 1; d < ND; ++d) mx = fmaxf(mx, lg[d]);
  float s = 0.0f;
  #pragma unroll
  for (int d = 0; d < ND; ++d) { lg[d] = expf(lg[d] - mx); s += lg[d]; }
  float inv = 1.0f / s;
  float* o = dp + (size_t)bv * ND * PP + p;
  #pragma unroll
  for (int d = 0; d < ND; ++d) o[(size_t)d * PP] = lg[d] * inv;
}

// ---------------- feature projection (1x1, 64->128) + BN + GELU
__global__ void k_featproj(const float* __restrict__ fm, const float* __restrict__ w,
                           const float* __restrict__ b,  const float* __restrict__ g,
                           const float* __restrict__ be, const float* __restrict__ m,
                           const float* __restrict__ v,  float* __restrict__ fp) {
  int tid = blockIdx.x * blockDim.x + threadIdx.x;
  if (tid >= NBV * BCH * PP) return;
  int p  = tid % PP;
  int c  = (tid / PP) % BCH;
  int bv = tid / (PP * BCH);
  const float* fmb = fm + (size_t)bv * FCH * PP + p;
  const float* wc  = w + (size_t)c * FCH;
  float acc = b[c];
  for (int ci = 0; ci < FCH; ++ci) acc = fmaf(wc[ci], fmb[(size_t)ci * PP], acc);
  fp[tid] = gelu_f(bn_f(acc, g[c], be[c], m[c], v[c]));
}

// ---------------- analytic 3x3 inverse of each K
__global__ void k_kinv(const float* __restrict__ K, float* __restrict__ kinv) {
  int bv = threadIdx.x;
  if (bv >= NBV) return;
  const float* k = K + bv * 9;
  float a = k[0], b = k[1], c = k[2], d = k[3], e = k[4], f = k[5], g = k[6], h = k[7], i = k[8];
  float A = e * i - f * h, Bc = f * g - d * i, C = d * h - e * g;
  float det = a * A + b * Bc + c * C;
  float inv = 1.0f / det;
  float* o = kinv + bv * 9;
  o[0] = A * inv;  o[1] = (c * h - b * i) * inv; o[2] = (b * f - c * e) * inv;
  o[3] = Bc * inv; o[4] = (a * i - c * g) * inv; o[5] = (c * d - a * f) * inv;
  o[6] = C * inv;  o[7] = (b * g - a * h) * inv; o[8] = (a * e - b * d) * inv;
}

__global__ void k_zero(float* __restrict__ p, int n) {
  int t = blockIdx.x * blockDim.x + threadIdx.x;
  if (t < n) p[t] = 0.0f;
}

// ---------------- lift + splat: one thread per (bv, pixel, depth), 128 atomics each
__global__ void k_scatter(const float* __restrict__ fp, const float* __restrict__ dp,
                          const float* __restrict__ kinv, const float* __restrict__ T,
                          const float* __restrict__ trust, float* __restrict__ bev) {
  int tid = blockIdx.x * blockDim.x + threadIdx.x;
  if (tid >= NBV * PP * ND) return;
  int d  = tid % ND;
  int p  = (tid / ND) % PP;
  int bv = tid / (ND * PP);
  int b  = bv / VV;
  float x = (float)(p % WFd), y = (float)(p / WFd);
  const float* Ki = kinv + bv * 9;
  float rx = Ki[0] * x + Ki[1] * y + Ki[2];
  float ry = Ki[3] * x + Ki[4] * y + Ki[5];
  float rz = Ki[6] * x + Ki[7] * y + Ki[8];
  float depth = D_MIN_ + (D_MAX_ - D_MIN_) * (float)d / (float)(ND - 1);
  float pcx = rx * depth, pcy = ry * depth, pcz = rz * depth;
  const float* Tm = T + bv * 16;   // row-major 4x4
  float xe = Tm[0] * pcx + Tm[1] * pcy + Tm[2] * pcz + Tm[3];
  float ye = Tm[4] * pcx + Tm[5] * pcy + Tm[6] * pcz + Tm[7];
  float u  = (xe + EXT_) / (2.0f * EXT_);
  float vq = (ye + EXT_) / (2.0f * EXT_);
  int col = (int)(u  * (float)(BW - 1));   // trunc-toward-zero matches astype(int32)
  int row = (int)(vq * (float)(BH - 1));
  if (col < 0 || col >= BW || row < 0 || row >= BH) return;
  float s = trust[bv] * dp[(size_t)bv * ND * PP + (size_t)d * PP + p];
  if (s == 0.0f) return;
  const float* fpb = fp + (size_t)bv * BCH * PP + p;
  float* bb = bev + (size_t)b * BCH * (BH * BW) + row * BW + col;
  for (int c = 0; c < BCH; ++c)
    atomicAdd(bb + (size_t)c * (BH * BW), fpb[(size_t)c * PP] * s);
}

// ---------------- f32 NCHW -> f16 NHWC repack (feeds WMMA B-matrix loads)
__global__ void k_nchw2nhwc_h(const float* __restrict__ src, _Float16* __restrict__ dst) {
  int tid = blockIdx.x * blockDim.x + threadIdx.x;
  if (tid >= BB * BH * BW * BCH) return;
  int c = tid % BCH;
  int x = (tid / BCH) % BW;
  int y = (tid / (BCH * BW)) % BH;
  int b = tid / (BCH * BW * BH);
  dst[tid] = (_Float16)src[(((size_t)b * BCH + c) * BH + y) * BW + x];
}

// ---------------- BEV 3x3 conv 128->128 via V_WMMA_F32_16X16X32_F16
// One wave computes a 16-outch x 16-pixel tile; K = 9 taps x 128 ci (4 chunks of 32).
// A-frag (16x32 f16): lane L -> row co_t*16 + (L&15); halves i: k=16*hl+i
//   -> two contiguous v8h loads from [tap][co][ci].
// B-frag (32x16 f16): lane L -> col pixel n=L&15; halves i: k=16*hl+i
//   -> one contiguous v16h load from NHWC f16 input.
__global__ void k_bev_conv_wmma(const _Float16* __restrict__ in, const _Float16* __restrict__ wA,
                                const float* __restrict__ cb, const float* __restrict__ g,
                                const float* __restrict__ be, const float* __restrict__ mu,
                                const float* __restrict__ var,
                                _Float16* __restrict__ out_h, float* __restrict__ out_f,
                                int write_f32) {
  int wave = (blockIdx.x * blockDim.x + threadIdx.x) >> 5;   // uniform per wave
  int lane = threadIdx.x & 31;
  if (wave >= BB * 8 * BH * 8) return;
  int xt   = wave & 7;
  int y    = (wave >> 3) & (BH - 1);
  int co_t = (wave >> 10) & 7;
  int b    = wave >> 13;
  int hl = lane >> 4;
  int n  = lane & 15;
  int xcol = xt * 16 + n;

  v8f acc = {};
  const _Float16* inb = in + (size_t)b * BH * BW * BCH;

  // speculative prefetch of the row below this tile (global_prefetch_b8)
  {
    int py = (y + 1 < BH) ? (y + 1) : y;
    __builtin_prefetch(inb + ((size_t)py * BW + xcol) * BCH, 0, 1);
  }

  #pragma unroll
  for (int ky = 0; ky < 3; ++ky) {
    int iy = y + ky - 1;
    #pragma unroll
    for (int kx = 0; kx < 3; ++kx) {
      int ix = xcol + kx - 1;
      bool bvalid = (iy >= 0) && (iy < BH) && (ix >= 0) && (ix < BW);
      const _Float16* bp = inb + ((size_t)iy * BW + ix) * BCH;
      int tap = ky * 3 + kx;
      const _Float16* ap = wA + ((size_t)tap * BCH + (co_t * 16 + n)) * BCH;
      #pragma unroll
      for (int ch = 0; ch < 4; ++ch) {
        v16h bmat = {};
        if (bvalid) bmat = *(const v16h*)(bp + ch * 32 + 16 * hl);
        union { v16h vv; v8h h2[2]; } am;
        am.h2[0] = *(const v8h*)(ap + ch * 32 + 16 * hl);
        am.h2[1] = *(const v8h*)(ap + ch * 32 + 16 * hl + 8);
        acc = __builtin_amdgcn_wmma_f32_16x16x32_f16(
            false, am.vv, false, bmat, (short)0, acc, false, false);
      }
    }
  }

  // fused bias + BN + GELU epilogue; D element j in lane -> co_local = j + 8*hl
  #pragma unroll
  for (int j = 0; j < 8; ++j) {
    int co = co_t * 16 + hl * 8 + j;
    float val = acc[j] + cb[co];
    val = gelu_f(bn_f(val, g[co], be[co], mu[co], var[co]));
    acc[j] = val;
  }
  if (write_f32) {
    #pragma unroll
    for (int j = 0; j < 8; ++j) {
      int co = co_t * 16 + hl * 8 + j;
      out_f[(((size_t)b * BCH + co) * BH + y) * BW + xcol] = acc[j];
    }
  } else {
    v8h ph;
    #pragma unroll
    for (int j = 0; j < 8; ++j) ph[j] = (_Float16)acc[j];
    *(v8h*)(out_h + (((size_t)b * BH + y) * BW + xcol) * BCH + co_t * 16 + hl * 8) = ph;
  }
}

extern "C" void kernel_launch(void* const* d_in, const int* in_sizes, int n_in,
                              void* d_out, int out_size, void* d_ws, size_t ws_size,
                              hipStream_t stream) {
  const float* feat  = (const float*)d_in[0];
  const float* Kl    = (const float*)d_in[1];
  const float* Tl    = (const float*)d_in[2];
  const float* trust = (const float*)d_in[3];
  const float* dh_w1 = (const float*)d_in[4];
  const float* dh_b1 = (const float*)d_in[5];
  const float* dh_g  = (const float*)d_in[6];
  const float* dh_be = (const float*)d_in[7];
  const float* dh_m  = (const float*)d_in[8];
  const float* dh_v  = (const float*)d_in[9];
  const float* dh_w2 = (const float*)d_in[10];
  const float* dh_b2 = (const float*)d_in[11];
  const float* fp_w  = (const float*)d_in[12];
  const float* fp_b  = (const float*)d_in[13];
  const float* fp_g  = (const float*)d_in[14];
  const float* fp_be = (const float*)d_in[15];
  const float* fp_m  = (const float*)d_in[16];
  const float* fp_v  = (const float*)d_in[17];
  const float* br_w1 = (const float*)d_in[18];
  const float* br_b1 = (const float*)d_in[19];
  const float* br_g1 = (const float*)d_in[20];
  const float* br_be1= (const float*)d_in[21];
  const float* br_m1 = (const float*)d_in[22];
  const float* br_v1 = (const float*)d_in[23];
  const float* br_w2 = (const float*)d_in[24];
  const float* br_b2 = (const float*)d_in[25];
  const float* br_g2 = (const float*)d_in[26];
  const float* br_be2= (const float*)d_in[27];
  const float* br_m2 = (const float*)d_in[28];
  const float* br_v2 = (const float*)d_in[29];

  // workspace carve-out (256B aligned slices)
  auto al = [](size_t x) { return (x + 255) & ~(size_t)255; };
  char* base = (char*)d_ws;
  float*    h    = (float*)base;    base += al((size_t)NBV * FCH * PP * sizeof(float));
  float*    dpb  = (float*)base;    base += al((size_t)NBV * ND  * PP * sizeof(float));
  float*    fpb  = (float*)base;    base += al((size_t)NBV * BCH * PP * sizeof(float));
  float*    kinv = (float*)base;    base += al((size_t)NBV * 9 * sizeof(float));
  float*    bev  = (float*)base;    base += al((size_t)BB * BCH * BH * BW * sizeof(float));
  _Float16* bevh = (_Float16*)base; base += al((size_t)BB * BH * BW * BCH * sizeof(_Float16));
  _Float16* h1h  = (_Float16*)base; base += al((size_t)BB * BH * BW * BCH * sizeof(_Float16));
  _Float16* wa1  = (_Float16*)base; base += al((size_t)9 * BCH * BCH * sizeof(_Float16));
  _Float16* wa2  = (_Float16*)base; base += al((size_t)9 * BCH * BCH * sizeof(_Float16));
  _Float16* fmh  = (_Float16*)base; base += al((size_t)NBV * PP * FCH * sizeof(_Float16));
  _Float16* wd1  = (_Float16*)base; base += al((size_t)9 * FCH * FCH * sizeof(_Float16));

  const int TB = 256;
  // weight repacks (BEV convs: 128x128, depth conv: 64x64)
  {
    int n = 9 * BCH * BCH;
    k_wcvt<<<(n + TB - 1) / TB, TB, 0, stream>>>(br_w1, wa1, BCH, BCH);
    k_wcvt<<<(n + TB - 1) / TB, TB, 0, stream>>>(br_w2, wa2, BCH, BCH);
    int n2 = 9 * FCH * FCH;
    k_wcvt<<<(n2 + TB - 1) / TB, TB, 0, stream>>>(dh_w1, wd1, FCH, FCH);
  }
  // feat_maps -> f16 channel-last
  {
    int n = NBV * PP * FCH;
    k_fm_nhwc<<<(n + TB - 1) / TB, TB, 0, stream>>>(feat, fmh);
  }
  // depth head conv1 via WMMA
  {
    int waves = NBV * 4 * PT16;   // 4224
    int blocks = (waves + 7) / 8;
    k_depth_conv1_wmma<<<blocks, TB, 0, stream>>>(fmh, wd1, dh_b1, dh_g, dh_be, dh_m, dh_v, h);
  }
  // depth probs
  {
    int n = NBV * PP;
    k_depth_softmax<<<(n + TB - 1) / TB, TB, 0, stream>>>(h, dh_w2, dh_b2, dpb);
  }
  // feature projection
  {
    int n = NBV * BCH * PP;
    k_featproj<<<(n + TB - 1) / TB, TB, 0, stream>>>(feat, fp_w, fp_b, fp_g, fp_be, fp_m, fp_v, fpb);
  }
  // K inverse
  k_kinv<<<1, 32, 0, stream>>>(Kl, kinv);
  // zero BEV accumulator
  {
    int n = BB * BCH * BH * BW;
    k_zero<<<(n + TB - 1) / TB, TB, 0, stream>>>(bev, n);
  }
  // lift + splat
  {
    int n = NBV * PP * ND;
    k_scatter<<<(n + TB - 1) / TB, TB, 0, stream>>>(fpb, dpb, kinv, Tl, trust, bev);
  }
  // repack BEV to f16 NHWC
  {
    int n = BB * BH * BW * BCH;
    k_nchw2nhwc_h<<<(n + TB - 1) / TB, TB, 0, stream>>>(bev, bevh);
  }
  // BEV refine conv1 (WMMA) -> f16 NHWC intermediate
  {
    int waves = BB * 8 * BH * 8;           // 16384 tiles, 8 waves per block
    int blocks = waves / 8;
    k_bev_conv_wmma<<<blocks, TB, 0, stream>>>(bevh, wa1, br_b1, br_g1, br_be1, br_m1, br_v1,
                                               h1h, nullptr, 0);
  }
  // BEV refine conv2 (WMMA) -> f32 NCHW output
  {
    int waves = BB * 8 * BH * 8;
    int blocks = waves / 8;
    k_bev_conv_wmma<<<blocks, TB, 0, stream>>>(h1h, wa2, br_b2, br_g2, br_be2, br_m2, br_v2,
                                               nullptr, (float*)d_out, 1);
  }
}
